// DecoderAttention_31860067401768
// MI455X (gfx1250) — matrix operand order repaired and checked by reading, not compiled, not run on hardware
//
#include <hip/hip_runtime.h>

// ---------------------------------------------------------------------------
// DecoderAttention on gfx1250 (MI455X): bf16 WMMA pipeline, f32 accumulate,
// async global->LDS operand staging (ASYNCcnt), double-buffered.
// ---------------------------------------------------------------------------

typedef __bf16 bf16_t;
typedef __attribute__((ext_vector_type(16))) __bf16       v16bf;
typedef __attribute__((ext_vector_type(8)))  float        v8f;
typedef __attribute__((ext_vector_type(4)))  unsigned int v4u;

#define VOCAB   32000
#define HIDDEN  1024
#define NB      4
#define DEC_LEN 512
#define ENC_LEN 1024
#define LDSROW  40   // 32 payload + 8 pad (bf16 elems) per tile row

struct Frag { union { v16bf bf; v4u u[2]; }; };

// ---------------------------------------------------------------------------
// Embedding: x[bs,h] = bf16(tok_emb[tok[bs]][h] + pos_emb[s][h])
// ---------------------------------------------------------------------------
__global__ __launch_bounds__(256) void k_embed(const int* __restrict__ tok,
                                               const float* __restrict__ te,
                                               const float* __restrict__ pe,
                                               bf16_t* __restrict__ x) {
  int bs = blockIdx.x;
  int s  = bs & (DEC_LEN - 1);
  int t  = tok[bs];
  const float* trow = te + (size_t)t * HIDDEN;
  const float* prow = pe + (size_t)s * HIDDEN;
  bf16_t* out = x + (size_t)bs * HIDDEN;
  for (int h = threadIdx.x; h < HIDDEN; h += 256)
    out[h] = (bf16_t)(trow[h] + prow[h]);
}

// f32 -> bf16 bulk convert
__global__ __launch_bounds__(256) void k_cvt(const float* __restrict__ in,
                                             bf16_t* __restrict__ out, long n) {
  long i = (long)blockIdx.x * 256 + threadIdx.x;
  long st = (long)gridDim.x * 256;
  for (; i < n; i += st) out[i] = (bf16_t)in[i];
}

// Transpose+convert f32[R,C] -> bf16[C,R]  (R,C multiples of 32)
__global__ __launch_bounds__(256) void k_trans_f32(const float* __restrict__ in,
                                                   bf16_t* __restrict__ out,
                                                   int R, int C) {
  __shared__ bf16_t tile[32][33];
  int c0 = blockIdx.x * 32, r0 = blockIdx.y * 32;
  for (int i = threadIdx.y; i < 32; i += 8)
    tile[i][threadIdx.x] = (bf16_t)in[(size_t)(r0 + i) * C + c0 + threadIdx.x];
  __syncthreads();
  for (int i = threadIdx.y; i < 32; i += 8)
    out[(size_t)(c0 + i) * R + r0 + threadIdx.x] = tile[threadIdx.x][i];
}

// Batched bf16[R,C] -> bf16[C,R] transpose
__global__ __launch_bounds__(256) void k_trans_bf16(const bf16_t* __restrict__ in,
                                                    bf16_t* __restrict__ out,
                                                    int R, int C, long sIn, long sOut) {
  in  += (size_t)blockIdx.z * sIn;
  out += (size_t)blockIdx.z * sOut;
  __shared__ bf16_t tile[32][33];
  int c0 = blockIdx.x * 32, r0 = blockIdx.y * 32;
  for (int i = threadIdx.y; i < 32; i += 8)
    tile[i][threadIdx.x] = in[(size_t)(r0 + i) * C + c0 + threadIdx.x];
  __syncthreads();
  for (int i = threadIdx.y; i < 32; i += 8)
    out[(size_t)(c0 + i) * R + r0 + threadIdx.x] = tile[threadIdx.x][i];
}

// ---------------------------------------------------------------------------
// WMMA GEMM:  C[M,N] = scale * (A[M,K] @ Bt[N,K]^T) + bias, optional causal
// mask, f32 or bf16 output.  Block = 8 waves -> 128x128 tile; each wave does
// a 32x64 tile = 2x4 V_WMMA_F32_16X16X32_BF16 accumulators.
// Operand tiles (A 128x32, Bt 128x32 bf16) are staged into LDS with
// global_load_async_to_lds_b128 (ASYNCcnt), double-buffered; fragments are
// read back with ds_load_b128 in the ISA per-lane layouts.
// Grid: (N/128, M/128, batches).  All dims divide evenly for this problem.
// ---------------------------------------------------------------------------
__global__ __launch_bounds__(256)
void k_gemm(const bf16_t* __restrict__ A, const bf16_t* __restrict__ Bt,
            const float* __restrict__ bias,
            float* __restrict__ Cf, bf16_t* __restrict__ Cb,
            int K, int lda, int ldb, int ldc,
            long sA, long sB, long sC,
            float scale, int causal) {
  __shared__ bf16_t smA[2][128][LDSROW];
  __shared__ bf16_t smB[2][128][LDSROW];

  A  += (size_t)blockIdx.z * sA;
  Bt += (size_t)blockIdx.z * sB;

  const int tid  = threadIdx.x;
  const int lane = tid & 31;
  const int wave = tid >> 5;
  const int mloc = (wave >> 1) * 32;   // 4 waves along M
  const int nloc = (wave & 1) * 64;    // 2 waves along N
  const int mblk = blockIdx.y * 128;
  const int nblk = blockIdx.x * 128;
  const int rsel = lane & 15;
  const int hi   = lane >> 4;

  // Cooperative staging: thread -> (row = tid>>2 [+64], col = (tid&3)*8 elems)
  const int srow = tid >> 2;
  const int scol = (tid & 3) * 8;
  const bf16_t* gA0 = A  + (size_t)(mblk + srow)      * lda + scol;
  const bf16_t* gA1 = A  + (size_t)(mblk + 64 + srow) * lda + scol;
  const bf16_t* gB0 = Bt + (size_t)(nblk + srow)      * ldb + scol;
  const bf16_t* gB1 = Bt + (size_t)(nblk + 64 + srow) * ldb + scol;

  auto stage = [&](int buf, int k0) {
    unsigned da0 = (unsigned)(size_t)&smA[buf][srow][scol];
    unsigned da1 = (unsigned)(size_t)&smA[buf][64 + srow][scol];
    unsigned db0 = (unsigned)(size_t)&smB[buf][srow][scol];
    unsigned db1 = (unsigned)(size_t)&smB[buf][64 + srow][scol];
    asm volatile("global_load_async_to_lds_b128 %0, %1, off"
                 :: "v"(da0), "v"((unsigned long long)(size_t)(gA0 + k0)) : "memory");
    asm volatile("global_load_async_to_lds_b128 %0, %1, off"
                 :: "v"(da1), "v"((unsigned long long)(size_t)(gA1 + k0)) : "memory");
    asm volatile("global_load_async_to_lds_b128 %0, %1, off"
                 :: "v"(db0), "v"((unsigned long long)(size_t)(gB0 + k0)) : "memory");
    asm volatile("global_load_async_to_lds_b128 %0, %1, off"
                 :: "v"(db1), "v"((unsigned long long)(size_t)(gB1 + k0)) : "memory");
  };

  v8f acc[2][4];
#pragma unroll
  for (int i = 0; i < 2; ++i)
#pragma unroll
    for (int j = 0; j < 4; ++j)
      acc[i][j] = (v8f){0.f, 0.f, 0.f, 0.f, 0.f, 0.f, 0.f, 0.f};

  stage(0, 0);  // prologue fill

  for (int k0 = 0; k0 < K; k0 += 32) {
    const int cur = (k0 >> 5) & 1;
    // our own async loads complete -> barrier makes everyone's complete
    asm volatile("s_wait_asynccnt 0x0" ::: "memory");
    __syncthreads();
    if (k0 + 32 < K) {
      stage(1 - cur, k0 + 32);
      __builtin_prefetch(gA0 + k0 + 64, 0, 3);  // warm L2/WGP$ ahead of async
      __builtin_prefetch(gB0 + k0 + 64, 0, 3);
    }

    Frag a[2], b[4];
#pragma unroll
    for (int i = 0; i < 2; ++i) {
      // A 16x32 layout: lanes0-15 K{0-7,16-23}, lanes16-31 K{8-15,24-31}
      const bf16_t* p = &smA[cur][mloc + i * 16 + rsel][hi * 8];
      a[i].u[0] = *(const v4u*)p;
      a[i].u[1] = *(const v4u*)(p + 16);
    }
#pragma unroll
    for (int j = 0; j < 4; ++j) {
      // B 32x16 layout: lanes0-15 K{0-7,16-23}, lanes16-31 K{8-15,24-31}
      const bf16_t* p = &smB[cur][nloc + j * 16 + rsel][hi * 8];
      b[j].u[0] = *(const v4u*)p;
      b[j].u[1] = *(const v4u*)(p + 16);
    }
#pragma unroll
    for (int i = 0; i < 2; ++i)
#pragma unroll
      for (int j = 0; j < 4; ++j)
        acc[i][j] = __builtin_amdgcn_wmma_f32_16x16x32_bf16(
            false, a[i].bf, false, b[j].bf, (short)0, acc[i][j], false, false);
  }

  // Epilogue: C/D f32 layout -> VGPR r holds rows {r, r+8}, lane&15 = column
#pragma unroll
  for (int i = 0; i < 2; ++i) {
#pragma unroll
    for (int j = 0; j < 4; ++j) {
      int col = nblk + nloc + j * 16 + rsel;
      float bv = bias ? bias[col] : 0.f;
#pragma unroll
      for (int r = 0; r < 8; ++r) {
        int row = mblk + mloc + i * 16 + (hi ? r + 8 : r);
        float v = acc[i][j][r] * scale + bv;
        if (causal && col > row) v = -1e30f;
        size_t idx = (size_t)row * ldc + col + (size_t)blockIdx.z * sC;
        if (Cb) Cb[idx] = (bf16_t)v;
        else    Cf[idx] = v;
      }
    }
  }
}

// ---------------------------------------------------------------------------
// Row softmax: f32 scores -> bf16 probabilities (one block per row, n<=1024)
// ---------------------------------------------------------------------------
__global__ __launch_bounds__(256) void k_softmax(const float* __restrict__ S,
                                                 bf16_t* __restrict__ P, int n) {
  __shared__ float red[256];
  long row = blockIdx.x;
  const float* s = S + row * (long)n;
  bf16_t* p = P + row * (long)n;
  int tid = threadIdx.x;

  float m = -3.4e38f;
  for (int c = tid; c < n; c += 256) m = fmaxf(m, s[c]);
  red[tid] = m; __syncthreads();
  for (int off = 128; off > 0; off >>= 1) {
    if (tid < off) red[tid] = fmaxf(red[tid], red[tid + off]);
    __syncthreads();
  }
  m = red[0]; __syncthreads();

  float ev[4]; int cnt = 0;
  float sum = 0.f;
  for (int c = tid; c < n; c += 256) { float e = __expf(s[c] - m); ev[cnt++] = e; sum += e; }
  red[tid] = sum; __syncthreads();
  for (int off = 128; off > 0; off >>= 1) {
    if (tid < off) red[tid] += red[tid + off];
    __syncthreads();
  }
  float inv = 1.f / red[0];
  cnt = 0;
  for (int c = tid; c < n; c += 256) p[c] = (bf16_t)(ev[cnt++] * inv);
}

// ---------------------------------------------------------------------------
extern "C" void kernel_launch(void* const* d_in, const int* in_sizes, int n_in,
                              void* d_out, int out_size, void* d_ws, size_t ws_size,
                              hipStream_t stream) {
  const float* enc     = (const float*)d_in[0];
  const int*   tok     = (const int*)d_in[1];
  const float* tok_emb = (const float*)d_in[2];
  const float* pos_emb = (const float*)d_in[3];
  const float* Wq_s = (const float*)d_in[4];  const float* bq_s = (const float*)d_in[5];
  const float* Wk_s = (const float*)d_in[6];  const float* bk_s = (const float*)d_in[7];
  const float* Wv_s = (const float*)d_in[8];  const float* bv_s = (const float*)d_in[9];
  const float* Wq_c = (const float*)d_in[10]; const float* bq_c = (const float*)d_in[11];
  const float* Wk_c = (const float*)d_in[12]; const float* bk_c = (const float*)d_in[13];
  const float* Wv_c = (const float*)d_in[14]; const float* bv_c = (const float*)d_in[15];
  const float* Wout = (const float*)d_in[16]; const float* bout = (const float*)d_in[17];
  float* out = (float*)d_out;

  const int H = HIDDEN, S = DEC_LEN, E = ENC_LEN;
  const int MS = NB * S;   // 2048 decoder rows
  const int ME = NB * E;   // 4096 encoder rows
  const float ascale = 0.03125f;  // 1/sqrt(1024)

  // -------- workspace layout (bf16 unless noted) --------
  unsigned char* w = (unsigned char*)d_ws;
  size_t off = 0;
  auto take = [&](size_t bytes) {
    void* p = w + off;
    off += (bytes + 255) & ~(size_t)255;
    return p;
  };
  bf16_t* x0    = (bf16_t*)take((size_t)MS * H * 2);          // embedded tokens
  bf16_t* encb  = (bf16_t*)take((size_t)ME * H * 2);          // encoder bf16
  bf16_t* wt    = (bf16_t*)take((size_t)H  * H * 2);          // reused W^T slot
  bf16_t* woutT = (bf16_t*)take((size_t)VOCAB * H * 2);       // Wout^T
  bf16_t* bufQ  = (bf16_t*)take((size_t)MS * H * 2);
  bf16_t* bufK  = (bf16_t*)take((size_t)ME * H * 2);
  bf16_t* bufV  = (bf16_t*)take((size_t)ME * H * 2);
  bf16_t* bufVT = (bf16_t*)take((size_t)ME * H * 2);
  float*  scr   = (float*) take((size_t)NB * S * E * 4);      // scores f32
  bf16_t* prob  = (bf16_t*)take((size_t)NB * S * E * 2);
  bf16_t* attn1 = (bf16_t*)take((size_t)MS * H * 2);
  bf16_t* out2  = (bf16_t*)take((size_t)MS * H * 2);

  dim3 tb(32, 8);
  dim3 gW(H / 32, H / 32);

  // -------- stage 0: embed + encoder convert --------
  k_embed<<<MS, 256, 0, stream>>>(tok, tok_emb, pos_emb, x0);
  k_cvt<<<2048, 256, 0, stream>>>(enc, encb, (long)ME * H);

  // -------- stage 1: self-attention --------
  k_trans_f32<<<gW, tb, 0, stream>>>(Wq_s, wt, H, H);
  k_gemm<<<dim3(H/128, MS/128, 1), 256, 0, stream>>>(x0, wt, bq_s, nullptr, bufQ,
      H, H, H, H, 0, 0, 0, 1.f, 0);
  k_trans_f32<<<gW, tb, 0, stream>>>(Wk_s, wt, H, H);
  k_gemm<<<dim3(H/128, MS/128, 1), 256, 0, stream>>>(x0, wt, bk_s, nullptr, bufK,
      H, H, H, H, 0, 0, 0, 1.f, 0);
  k_trans_f32<<<gW, tb, 0, stream>>>(Wv_s, wt, H, H);
  k_gemm<<<dim3(H/128, MS/128, 1), 256, 0, stream>>>(x0, wt, bv_s, nullptr, bufV,
      H, H, H, H, 0, 0, 0, 1.f, 0);
  // V^T per batch: [S,H] -> [H,S]
  k_trans_bf16<<<dim3(H/32, S/32, NB), tb, 0, stream>>>(bufV, bufVT, S, H,
      (long)S * H, (long)H * S);
  // scores = scale * Q K^T with causal mask   [per batch: 512x512, K=1024]
  k_gemm<<<dim3(S/128, S/128, NB), 256, 0, stream>>>(bufQ, bufK, nullptr, scr, nullptr,
      H, H, H, S, (long)S * H, (long)S * H, (long)S * S, ascale, 1);
  k_softmax<<<NB * S, 256, 0, stream>>>(scr, prob, S);
  // attn1 = P @ V  (Bt = V^T rows)   [per batch: 512x1024, K=512]
  k_gemm<<<dim3(H/128, S/128, NB), 256, 0, stream>>>(prob, bufVT, nullptr, nullptr, attn1,
      S, S, S, H, (long)S * S, (long)H * S, (long)S * H, 1.f, 0);

  // -------- stage 2: cross-attention --------
  k_trans_f32<<<gW, tb, 0, stream>>>(Wq_c, wt, H, H);
  k_gemm<<<dim3(H/128, MS/128, 1), 256, 0, stream>>>(attn1, wt, bq_c, nullptr, bufQ,
      H, H, H, H, 0, 0, 0, 1.f, 0);
  k_trans_f32<<<gW, tb, 0, stream>>>(Wk_c, wt, H, H);
  k_gemm<<<dim3(H/128, ME/128, 1), 256, 0, stream>>>(encb, wt, bk_c, nullptr, bufK,
      H, H, H, H, 0, 0, 0, 1.f, 0);
  k_trans_f32<<<gW, tb, 0, stream>>>(Wv_c, wt, H, H);
  k_gemm<<<dim3(H/128, ME/128, 1), 256, 0, stream>>>(encb, wt, bv_c, nullptr, bufV,
      H, H, H, H, 0, 0, 0, 1.f, 0);
  k_trans_bf16<<<dim3(H/32, E/32, NB), tb, 0, stream>>>(bufV, bufVT, E, H,
      (long)E * H, (long)H * E);
  // scores2 = scale * Qc Kc^T   [per batch: 512x1024, K=1024]
  k_gemm<<<dim3(E/128, S/128, NB), 256, 0, stream>>>(bufQ, bufK, nullptr, scr, nullptr,
      H, H, H, E, (long)S * H, (long)E * H, (long)S * E, ascale, 0);
  k_softmax<<<NB * S, 256, 0, stream>>>(scr, prob, E);
  // out2 = P2 @ Vc   [per batch: 512x1024, K=1024]
  k_gemm<<<dim3(H/128, S/128, NB), 256, 0, stream>>>(prob, bufVT, nullptr, nullptr, out2,
      E, E, E, H, (long)S * E, (long)H * E, (long)S * H, 1.f, 0);

  // -------- stage 3: vocab projection (dominant GEMM) --------
  k_trans_f32<<<dim3(VOCAB/32, H/32), tb, 0, stream>>>(Wout, woutT, H, VOCAB);
  k_gemm<<<dim3(VOCAB/128, MS/128, 1), 256, 0, stream>>>(out2, woutT, bout, out, nullptr,
      H, H, H, VOCAB, 0, 0, 0, 1.f, 0);
}
